// Decoder_76836964926387
// MI455X (gfx1250) — matrix operand order repaired and checked by reading, not compiled, not run on hardware
//
#include <hip/hip_runtime.h>
#include <hip/hip_bf16.h>

// ---------------------------------------------------------------------------
// Decoder pipeline for MI455X (gfx1250), wave32.
// Heavy 219x219x27 convs -> implicit GEMM: TDM (tensor_load_to_lds) stages
// A/B tiles into triple-buffered LDS (HW pad gives conflict-free stride 40),
// v_wmma_f32_16x16x32_f16 computes, s_wait_tensorcnt pipelines 2 tiles deep.
// Instance-norm follows upconv/c1/c2, so their biases cancel and are skipped.
// ---------------------------------------------------------------------------

typedef __attribute__((ext_vector_type(16))) _Float16 v16h;
typedef __attribute__((ext_vector_type(8)))  _Float16 v8h;
typedef __attribute__((ext_vector_type(8)))  float    v8f;
typedef __attribute__((ext_vector_type(4)))  unsigned int v4u;
typedef __attribute__((ext_vector_type(8)))  int v8i;
typedef __attribute__((ext_vector_type(4)))  int v4i;

#define CC   219            // real channels of concat tensor
#define CP   224            // channel pad (multiple of 32)
#define KT   27             // 3x3x3 taps
#define KTOT (KT*CP)        // 6048 = GEMM K
#define NP   256            // padded output-channel rows
#define NVOX 32768          // 32^3
#define SXH  CP             // halo-buffer strides (halves)
#define SYH  (34*CP)
#define SZH  (34*34*CP)

// GEMM tiling
#define BM 64
#define BN 128
#define BK 32
#define LDA 40              // LDS row stride (halves): 32 data + 8 pad (TDM pad)
#define NKT (KT*(CP/BK))    // 189 K-tiles
#define ABUF (BM*LDA)       // halves per A buffer (2560)
#define BBUF (BN*LDA)       // halves per B buffer (5120)

#ifdef __AMDGCN__
#if defined(__clang_major__) && __clang_major__ >= 23
#define TDM_LOAD(g0, g1, g2, g3) \
    __builtin_amdgcn_tensor_load_to_lds((g0), (g1), (g2), (g3), (v8i)(0), 0)
#else
#define TDM_LOAD(g0, g1, g2, g3) \
    __builtin_amdgcn_tensor_load_to_lds((g0), (g1), (g2), (g3), 0)
#endif
#endif

// ---------------- transposed conv 4^3 stride2 (411ch 16^3 -> 64ch 32^3) -----
__global__ __launch_bounds__(256)
void upconv_kernel(const float* __restrict__ Cin, const float* __restrict__ w,
                   float* __restrict__ T)
{
    const int v  = blockIdx.x * 256 + threadIdx.x;   // 0..32767
    const int co = blockIdx.y;                        // 0..63
    const int x = v & 31, y = (v >> 5) & 31, z = v >> 10;

    int zi[2], zk[2], nz = 0;
    int yi[2], yk[2], ny = 0;
    int xi[2], xk[2], nx = 0;
    #pragma unroll
    for (int k = 0; k < 4; ++k) {
        int t;
        t = z + 1 - k; if (t >= 0 && !(t & 1) && (t >> 1) < 16) { zi[nz] = t >> 1; zk[nz] = k; ++nz; }
        t = y + 1 - k; if (t >= 0 && !(t & 1) && (t >> 1) < 16) { yi[ny] = t >> 1; yk[ny] = k; ++ny; }
        t = x + 1 - k; if (t >= 0 && !(t & 1) && (t >> 1) < 16) { xi[nx] = t >> 1; xk[nx] = k; ++nx; }
    }

    float acc = 0.f;
    for (int ci = 0; ci < 411; ++ci) {
        const float* Cc = Cin + ((size_t)ci << 12);                 // 16^3
        const float* wc = w + (((size_t)ci * 64 + co) << 6);        // 4^3
        for (int a = 0; a < nz; ++a) {
            const float* Cz = Cc + (zi[a] << 8);
            const float* wz = wc + zk[a] * 16;
            for (int b = 0; b < ny; ++b) {
                const float* Cy = Cz + (yi[b] << 4);
                const float* wy = wz + yk[b] * 4;
                for (int c = 0; c < nx; ++c)
                    acc += Cy[xi[c]] * wy[xk[c]];
            }
        }
    }
    T[((size_t)co << 15) + v] = acc;   // bias skipped: IN cancels it
}

// ---------------- per-channel mean / inv-std over 32^3 ----------------------
__global__ __launch_bounds__(256)
void stats_kernel(const float* __restrict__ X, float* __restrict__ st)
{
    const int c = blockIdx.x;
    const float* p = X + ((size_t)c << 15);
    float s = 0.f, s2 = 0.f;
    for (int i = threadIdx.x; i < NVOX; i += 256) {
        float v = p[i]; s += v; s2 += v * v;
    }
    __shared__ float sh[512];
    sh[threadIdx.x] = s; sh[256 + threadIdx.x] = s2;
    __syncthreads();
    for (int off = 128; off > 0; off >>= 1) {
        if (threadIdx.x < off) {
            sh[threadIdx.x]       += sh[threadIdx.x + off];
            sh[256 + threadIdx.x] += sh[256 + threadIdx.x + off];
        }
        __syncthreads();
    }
    if (threadIdx.x == 0) {
        float m   = sh[0]   * (1.f / NVOX);
        float var = sh[256] * (1.f / NVOX) - m * m;
        st[2 * c]     = m;
        st[2 * c + 1] = rsqrtf(var + 1e-5f);
    }
}

// ---------------- 27-shift correlation volume -------------------------------
__global__ __launch_bounds__(256)
void corr_kernel(const float* __restrict__ tgt, const float* __restrict__ src,
                 float* __restrict__ cost)
{
    const int v = blockIdx.x * 256 + threadIdx.x;
    const int j = blockIdx.y;                       // 0..26
    const int dz = j / 9 - 1, dy = (j / 3) % 3 - 1, dx = j % 3 - 1;
    const int x = v & 31, y = (v >> 5) & 31, z = v >> 10;
    const int zz = z + dz, yy = y + dy, xx = x + dx;
    float s = 0.f;
    if ((unsigned)zz <= 31u && (unsigned)yy <= 31u && (unsigned)xx <= 31u) {
        const int vs = (zz << 10) | (yy << 5) | xx;
        for (int c = 0; c < 64; ++c)
            s += tgt[((size_t)c << 15) + v] * src[((size_t)c << 15) + vs];
    }
    cost[((size_t)j << 15) + v] = s * (1.f / 64.f);
}

// ---------------- build f16 channels-last halo buffer (concat stage) --------
__global__ __launch_bounds__(CP)
void pack1_kernel(const float* __restrict__ tgt, const float* __restrict__ src,
                  const float* __restrict__ cost, const float* __restrict__ traw,
                  const float* __restrict__ stA, _Float16* __restrict__ A16)
{
    const int p = blockIdx.x;                 // 0..34^3-1
    const int c = threadIdx.x;                // 0..223
    const int xp = p % 34, t = p / 34, yp = t % 34, zp = t / 34;
    float val = 0.f;
    if (xp >= 1 && xp <= 32 && yp >= 1 && yp <= 32 && zp >= 1 && zp <= 32 && c < CC) {
        const int v = ((zp - 1) << 10) | ((yp - 1) << 5) | (xp - 1);
        if (c < 64)       val = tgt[((size_t)c << 15) + v];
        else if (c < 128) val = src[((size_t)(c - 64) << 15) + v];
        else if (c < 155) val = cost[((size_t)(c - 128) << 15) + v];
        else {
            const int cc = c - 155;
            float r  = traw[((size_t)cc << 15) + v];
            float nv = (r - stA[2 * cc]) * stA[2 * cc + 1];
            val = nv >= 0.f ? nv : 0.1f * nv;
        }
    }
    A16[(size_t)p * CP + c] = (_Float16)val;
}

// ---------------- norm+leaky conv output -> f16 halo buffer -----------------
__global__ __launch_bounds__(CP)
void pack2_kernel(const float* __restrict__ Y, const float* __restrict__ st,
                  _Float16* __restrict__ A16)
{
    const int p = blockIdx.x;
    const int c = threadIdx.x;
    const int xp = p % 34, t = p / 34, yp = t % 34, zp = t / 34;
    float val = 0.f;
    if (xp >= 1 && xp <= 32 && yp >= 1 && yp <= 32 && zp >= 1 && zp <= 32 && c < CC) {
        const int v = ((zp - 1) << 10) | ((yp - 1) << 5) | (xp - 1);
        float r  = Y[((size_t)c << 15) + v];
        float nv = (r - st[2 * c]) * st[2 * c + 1];
        val = nv >= 0.f ? nv : 0.1f * nv;
    }
    A16[(size_t)p * CP + c] = (_Float16)val;
}

// ---------------- norm+leaky -> f32 Cn (first output) -----------------------
__global__ __launch_bounds__(256)
void pack3_kernel(const float* __restrict__ Y, const float* __restrict__ st,
                  float* __restrict__ Cn)
{
    const int idx = blockIdx.x * 256 + threadIdx.x;   // < 219*32768
    const int c = idx >> 15;
    float r  = Y[idx];
    float nv = (r - st[2 * c]) * st[2 * c + 1];
    Cn[idx] = nv >= 0.f ? nv : 0.1f * nv;
}

// ---------------- weights OIDHW f32 -> f16 [co][tap][ci] (K contiguous) -----
__global__ __launch_bounds__(256)
void wconv_kernel(const float* __restrict__ w, _Float16* __restrict__ W16)
{
    const int idx = blockIdx.x * 256 + threadIdx.x;   // < NP*KTOT
    const int co = idx / KTOT;
    const int k  = idx % KTOT;
    const int kt = k / CP, ci = k % CP;
    float val = 0.f;
    if (co < CC && ci < CC)
        val = w[((size_t)(co * CC + ci)) * KT + kt];
    W16[idx] = (_Float16)val;
}

// ---------------- WMMA implicit-GEMM conv, TDM-staged -----------------------
// Y[n][m] = sum_k A[m][k] * W[n][k].  BM=64 voxels x BN=128 out-channels.
// 8 waves: wm=wid&3 (M 16-tile), wn=wid>>2 (N 64-half) -> 4 accumulators.
// Triple-buffered LDS; wave0 issues 2 TDM loads per K-tile, 2 tiles ahead;
// issue-side addresses are maintained incrementally (no div/mod in the loop);
// completion via s_wait_tensorcnt (harmless no-op on waves 1-7);
// one workgroup barrier per K-tile.
__global__ __launch_bounds__(256)
void gemm_conv3d(const _Float16* __restrict__ A, const _Float16* __restrict__ W,
                 float* __restrict__ Y)
{
    __shared__ _Float16 sA[3][ABUF];
    __shared__ _Float16 sB[3][BBUF];

    const int tid  = threadIdx.x;
    const int v0   = blockIdx.x * BM;          // voxel tile base (2 x-rows)
    const int co0  = blockIdx.y * BN;          // out-channel tile base
    const int z0   = v0 >> 10;
    const int y0   = (v0 >> 5) & 31;

    const int lane = tid & 31;
    const int wid  = tid >> 5;
    const int wm   = wid & 3;                  // 4 M-tiles of 16
    const int wn   = wid >> 2;                 // 2 N-halves of 64

    // fragment LDS offsets (halves), per ISA 16-bit A/B VGPR layouts
    const int fr  = lane & 15;
    const int akb = (lane < 16) ? 0 : 8;
    const int bkb = (lane < 16) ? 0 : 8;
    const int aoff = (wm * 16 + fr) * LDA + akb;
    int boff[4];
    #pragma unroll
    for (int j = 0; j < 4; ++j) boff[j] = (wn * 64 + 16 * j + fr) * LDA + bkb;

    v8f acc[4] = {};

    // ---- TDM descriptors (wave-uniform; SGPR resident) ----
    const unsigned ldsA0 = (unsigned)(uintptr_t)&sA[0][0];
    const unsigned ldsB0 = (unsigned)(uintptr_t)&sB[0][0];
    // group1 common: data_size=2B(code1), pad_enable, pad_interval=16dw(code3),
    // pad_amount=4dw(code3); tensor dims huge (no OOB clamp; reads in-bounds).
    const int G1C0 = (1 << 16) | (1 << 20) | (3 << 22) | (3 << 25);
    const v8i g1a = { G1C0, 0, 0x4000, 0x4000 | (32 << 16),
                      32 | (2 << 16),          // A: tile_dim1=32 (x), tile_dim2=2 (y)
                      SXH, (SYH & 0xFFFF) << 16, 0 };
    const v8i g1b = { G1C0, 0, 0x4000, 0x4000 | (32 << 16),
                      BN,                      // B: tile_dim1=128 rows, tile_dim2=0
                      KTOT, 0, 0 };
    const v4i g2 = { 0x40000000, 0x40000000, 0, 0 };   // dim2/dim3 huge, stride2=0
    const v4i g3 = { 0, 0, 0, 0 };

    // ---- issue-side state (meaningful on wave0 only), updated incrementally
    const _Float16* anext = A + (size_t)z0 * SZH + (size_t)y0 * SYH;  // tile 0
    const _Float16* bnext = W + (size_t)co0 * KTOT;
    int cx = 0, tx = 0, ty = 0;   // ci-chunk (0..6), dx (0..2), dy (0..2)
    unsigned ibufA = ldsA0, ibufB = ldsB0;   // LDS dest, rotates over 3 buffers
    int ibuf = 0;

    auto issue = [&]() {
#ifdef __AMDGCN__
        unsigned long long ga = (unsigned long long)(uintptr_t)anext;
        v4u g0a = { 1u, ibufA, (unsigned)ga,
                    (unsigned)((ga >> 32) & 0x01FFFFFFu) | (2u << 30) };
        TDM_LOAD(g0a, g1a, g2, g3);
        unsigned long long gb = (unsigned long long)(uintptr_t)bnext;
        v4u g0b = { 1u, ibufB, (unsigned)gb,
                    (unsigned)((gb >> 32) & 0x01FFFFFFu) | (2u << 30) };
        TDM_LOAD(g0b, g1b, g2, g3);
#endif
        // advance B: K is contiguous (+BK halves per tile)
        bnext += BK;
        // advance A: ci chunk -> dx -> dy -> dz walk, incremental pointer
        if (++cx == CP / BK) {
            cx = 0;
            if (++tx == 3) {
                tx = 0;
                if (++ty == 3) { ty = 0; anext += SZH - 2 * SYH - 2 * SXH - (CP - BK); }
                else           {          anext += SYH - 2 * SXH - (CP - BK); }
            } else             {          anext += SXH - (CP - BK); }
        } else                 {          anext += BK; }
        // rotate destination buffers
        if (ibuf == 2) { ibuf = 0; ibufA = ldsA0;            ibufB = ldsB0; }
        else           { ++ibuf;   ibufA += (unsigned)(ABUF * 2); ibufB += (unsigned)(BBUF * 2); }
    };

    const bool w0 = (wid == 0);
    if (w0) { issue(); issue(); }              // tiles 0,1 -> buffers 0,1

    int buf = 0;
    for (int k = 0; k < NKT - 1; ++k) {
#ifdef __AMDGCN__
        __builtin_amdgcn_s_wait_tensorcnt((short)2);   // tile k resident (wave0)
#endif
        __syncthreads();                               // visible to all waves
        if (w0 && k + 2 < NKT) issue();                // 2 tiles ahead

        const _Float16* aB = &sA[buf][0];
        const _Float16* bB = &sB[buf][0];
        v8h a_lo = *(const v8h*)(aB + aoff);
        v8h a_hi = *(const v8h*)(aB + aoff + 16);
        v16h afrag = __builtin_shufflevector(a_lo, a_hi,
            0,1,2,3,4,5,6,7,8,9,10,11,12,13,14,15);
        #pragma unroll
        for (int j = 0; j < 4; ++j) {
            v8h b_lo = *(const v8h*)(bB + boff[j]);
            v8h b_hi = *(const v8h*)(bB + boff[j] + 16);
            v16h bfrag = __builtin_shufflevector(b_lo, b_hi,
                0,1,2,3,4,5,6,7,8,9,10,11,12,13,14,15);
#ifdef __AMDGCN__
            acc[j] = __builtin_amdgcn_wmma_f32_16x16x32_f16(
                false, afrag, false, bfrag, (short)0, acc[j], false, false);
#else
            (void)afrag; (void)bfrag;
#endif
        }
        buf = (buf == 2) ? 0 : buf + 1;
    }
    // peeled last K-tile: drain TENSORcnt fully
#ifdef __AMDGCN__
    __builtin_amdgcn_s_wait_tensorcnt((short)0);
#endif
    __syncthreads();
    {
        const _Float16* aB = &sA[buf][0];
        const _Float16* bB = &sB[buf][0];
        v8h a_lo = *(const v8h*)(aB + aoff);
        v8h a_hi = *(const v8h*)(aB + aoff + 16);
        v16h afrag = __builtin_shufflevector(a_lo, a_hi,
            0,1,2,3,4,5,6,7,8,9,10,11,12,13,14,15);
        #pragma unroll
        for (int j = 0; j < 4; ++j) {
            v8h b_lo = *(const v8h*)(bB + boff[j]);
            v8h b_hi = *(const v8h*)(bB + boff[j] + 16);
            v16h bfrag = __builtin_shufflevector(b_lo, b_hi,
                0,1,2,3,4,5,6,7,8,9,10,11,12,13,14,15);
#ifdef __AMDGCN__
            acc[j] = __builtin_amdgcn_wmma_f32_16x16x32_f16(
                false, afrag, false, bfrag, (short)0, acc[j], false, false);
#else
            (void)afrag; (void)bfrag;
#endif
        }
    }

    // store: D 16x16 f32 layout -> M = r + (lane<16?0:8), N = lane&15
    const int sm = wm * 16 + ((lane < 16) ? 0 : 8);
    const int n0 = co0 + wn * 64 + (lane & 15);
    #pragma unroll
    for (int j = 0; j < 4; ++j) {
        float* yp = Y + (((size_t)(n0 + 16 * j)) << 15) + v0 + sm;
        #pragma unroll
        for (int r = 0; r < 8; ++r) yp[r] = acc[j][r];
    }
}

// ---------------- 3-channel output conv (tiny, direct) ----------------------
__global__ __launch_bounds__(256)
void outconv_kernel(const float* __restrict__ Cn, const float* __restrict__ ow,
                    const float* __restrict__ ob, float* __restrict__ out)
{
    const int v  = blockIdx.x * 256 + threadIdx.x;   // exact 32768
    const int oc = blockIdx.y;                        // 0..2
    const int x = v & 31, y = (v >> 5) & 31, z = v >> 10;
    float acc = ob[oc];
    for (int kt = 0; kt < KT; ++kt) {
        const int dz = kt / 9 - 1, dy = (kt / 3) % 3 - 1, dx = kt % 3 - 1;
        const int zz = z + dz, yy = y + dy, xx = x + dx;
        if ((unsigned)zz > 31u || (unsigned)yy > 31u || (unsigned)xx > 31u) continue;
        const int vv = (zz << 10) | (yy << 5) | xx;
        const float* wp = ow + (size_t)oc * CC * KT + kt;
        const float* cp = Cn + vv;
        for (int ci = 0; ci < CC; ++ci)
            acc += cp[(size_t)ci << 15] * wp[ci * KT];
    }
    out[((size_t)oc << 15) + v] = acc;
}

// ---------------------------------------------------------------------------
extern "C" void kernel_launch(void* const* d_in, const int* in_sizes, int n_in,
                              void* d_out, int out_size, void* d_ws, size_t ws_size,
                              hipStream_t stream)
{
    (void)in_sizes; (void)n_in; (void)out_size; (void)ws_size;
    const float* src   = (const float*)d_in[0];
    const float* tgt   = (const float*)d_in[1];
    const float* C     = (const float*)d_in[2];
    const float* up_w  = (const float*)d_in[3];
    /* up_b d_in[4], c1_b d_in[6], c2_b d_in[8]: cancelled by instance norm */
    const float* c1_w  = (const float*)d_in[5];
    const float* c2_w  = (const float*)d_in[7];
    const float* out_w = (const float*)d_in[9];
    const float* out_b = (const float*)d_in[10];

    char* ws = (char*)d_ws;
    const size_t szA16 = (size_t)34 * 34 * 34 * CP * sizeof(_Float16); // 17,608,192
    const size_t szW16 = (size_t)NP * KTOT * sizeof(_Float16);         //  3,096,576
    const size_t szY   = (size_t)NP * NVOX * sizeof(float);            // 33,554,432
    _Float16* A16 = (_Float16*)ws;
    _Float16* W16 = (_Float16*)(ws + szA16);
    float*    Yb  = (float*)(ws + szA16 + szW16);
    float*    Traw = Yb;                                   // dead before GEMM
    float*    Cost = (float*)((char*)Yb + (8u << 20));     // dead before GEMM
    float*    st   = (float*)(ws + szA16 + szW16 + szY);   // 1536 floats
    float* stA = st, *stB = st + 512, *stC = st + 1024;

    float* Cn  = (float*)d_out;                  // 219*32768 floats
    float* Out = Cn + (size_t)CC * NVOX;         // 3*32768 floats

    // 1) transposed conv -> raw 64x32^3, then its IN stats
    upconv_kernel<<<dim3(128, 64), 256, 0, stream>>>(C, up_w, Traw);
    stats_kernel <<<64, 256, 0, stream>>>(Traw, stA);
    // 2) correlation volume
    corr_kernel  <<<dim3(128, 27), 256, 0, stream>>>(tgt, src, Cost);
    // 3) concat -> f16 channels-last halo buffer
    pack1_kernel <<<34 * 34 * 34, CP, 0, stream>>>(tgt, src, Cost, Traw, stA, A16);
    // 4) conv1: weights -> f16, WMMA GEMM, stats, repack
    wconv_kernel <<<(NP * KTOT) / 256, 256, 0, stream>>>(c1_w, W16);
    gemm_conv3d  <<<dim3(NVOX / BM, NP / BN), 256, 0, stream>>>(A16, W16, Yb);
    stats_kernel <<<CC, 256, 0, stream>>>(Yb, stB);
    pack2_kernel <<<34 * 34 * 34, CP, 0, stream>>>(Yb, stB, A16);
    // 5) conv2: same machinery
    wconv_kernel <<<(NP * KTOT) / 256, 256, 0, stream>>>(c2_w, W16);
    gemm_conv3d  <<<dim3(NVOX / BM, NP / BN), 256, 0, stream>>>(A16, W16, Yb);
    stats_kernel <<<CC, 256, 0, stream>>>(Yb, stC);
    // 6) Cn = leaky(IN(conv2)) -> output #1
    pack3_kernel <<<(CC * NVOX) / 256, 256, 0, stream>>>(Yb, stC, Cn);
    // 7) tiny 3-channel output conv -> output #2
    outconv_kernel<<<dim3(128, 3), 256, 0, stream>>>(Cn, out_w, out_b, Out);
}
